// FlowEstimator3D_74552042324064
// MI455X (gfx1250) — matrix operand order repaired and checked by reading, not compiled, not run on hardware
//
#include <hip/hip_runtime.h>
#include <hip/hip_bf16.h>

typedef __bf16 v16bf __attribute__((ext_vector_type(16)));
typedef float  v8f   __attribute__((ext_vector_type(8)));

#define NN 8192   // points per batch
#define BB 4      // batch
#define KK 16     // knn neighbors
#define WN 8      // weight-net hidden channels

__device__ __forceinline__ float  leakyf(float x) { return x >= 0.0f ? x : 0.1f * x; }
__device__ __forceinline__ __bf16 tobf(float x)   { return (__bf16)x; }

union FragBF {
  v16bf v;
  uint4 q[2];
};

// ---------------- fp32 -> bf16 weight convert with zero K-padding ----------
__global__ void cvt_pad_bf16(const float* __restrict__ src, __bf16* __restrict__ dst,
                             int rows, int ksrc, int kdst) {
  int total = rows * kdst;
  for (int i = blockIdx.x * blockDim.x + threadIdx.x; i < total;
       i += gridDim.x * blockDim.x) {
    int r = i / kdst;
    int k = i - r * kdst;
    dst[i] = (k < ksrc) ? tobf(src[(size_t)r * ksrc + k]) : tobf(0.0f);
  }
}

// ---------------- fused point-conv layer -----------------------------------
// One 256-thread block (8 waves) handles a tile of 16 points:
//   phase 1 (VALU): gather neighbors, weight-net, agg[16][CP3*8] -> LDS bf16
//   phase 2 (WMMA): D[16 pts x 128 out] = agg(16 x Fpad) * Wlin^T, bf16->f32
__global__ __launch_bounds__(256) void point_conv_kernel(
    const float* __restrict__ xyz, const float* __restrict__ featIn,
    const int* __restrict__ knn, const float* __restrict__ w_wn,
    const float* __restrict__ b_wn, const __bf16* __restrict__ wbf,
    const float* __restrict__ b_lin, float* __restrict__ outF,
    int Cin, int Fpad) {
  extern __shared__ char smem[];
  int*    idxL  = (int*)smem;                        // 16*16 ints     (1 KB)
  float*  wmatL = (float*)(smem + 1024);             // 16*16*8 fp32   (8 KB)
  __bf16* aggL  = (__bf16*)(smem + 1024 + 8192);     // 16*Fpad bf16

  const int t     = threadIdx.x;
  const int tile  = blockIdx.x;
  const int b     = tile / (NN / 16);
  const int tileN = (tile % (NN / 16)) * 16;
  const int CP3   = Cin + 3;
  const int Cout  = 128;

  // Warm the L2-resident weight panel (global_prefetch_b8)
  if (t < 128) __builtin_prefetch(wbf + (size_t)t * Fpad, 0, 1);

  const int p = t >> 4;   // point in tile  (0..15)
  const int s = t & 15;   // lane-in-point  (0..15)
  const int n = tileN + p;

  // knn indices for this tile
  idxL[p * KK + s] = knn[((size_t)b * NN + n) * KK + s];
  __syncthreads();

  // weight-net: column k=s of w-matrix for point p
  {
    int   j  = idxL[p * KK + s];
    float px = xyz[((size_t)b * 3 + 0) * NN + j] - xyz[((size_t)b * 3 + 0) * NN + n];
    float py = xyz[((size_t)b * 3 + 1) * NN + j] - xyz[((size_t)b * 3 + 1) * NN + n];
    float pz = xyz[((size_t)b * 3 + 2) * NN + j] - xyz[((size_t)b * 3 + 2) * NN + n];
#pragma unroll
    for (int m = 0; m < WN; ++m) {
      float v = w_wn[m * 3 + 0] * px + w_wn[m * 3 + 1] * py +
                w_wn[m * 3 + 2] * pz + b_wn[m];
      wmatL[(p * KK + s) * WN + m] = leakyf(v);
    }
  }
  __syncthreads();

  // agg[c][m] = sum_k f[c][idx_k] * w[m][k]   (f = concat(xyz, feat))
  for (int c = s; c < CP3; c += 16) {
    float acc[WN];
#pragma unroll
    for (int m = 0; m < WN; ++m) acc[m] = 0.0f;
    for (int k = 0; k < KK; ++k) {
      int   j   = idxL[p * KK + k];
      float val = (c < 3) ? xyz[((size_t)b * 3 + c) * NN + j]
                          : featIn[((size_t)b * Cin + (c - 3)) * NN + j];
      const float* wm = &wmatL[(p * KK + k) * WN];
#pragma unroll
      for (int m = 0; m < WN; ++m) acc[m] = fmaf(val, wm[m], acc[m]);
    }
#pragma unroll
    for (int m = 0; m < WN; ++m) aggL[p * Fpad + c * WN + m] = tobf(acc[m]);
  }
  if (s == 0) {  // zero the K padding
    for (int q = CP3 * WN; q < Fpad; ++q) aggL[p * Fpad + q] = tobf(0.0f);
  }
  __syncthreads();

  // WMMA GEMM: each wave owns one 16x16 output tile (8 waves -> 128 channels)
  const int lane = t & 31;
  const int wid  = t >> 5;
  const int rc   = lane & 15;   // A-row / B&D-column within tile
  const int sel  = lane >> 4;
  const int o    = wid * 16 + rc;

  v8f acc = {};
  const __bf16* arow = aggL + rc * Fpad;
  const __bf16* bcol = wbf + (size_t)o * Fpad;
  const int nKB = Fpad >> 5;
  for (int kb = 0; kb < nKB; ++kb) {
    FragBF A, Bm;
    const __bf16* ap = arow + kb * 32 + sel * 8;     // K {base..+7, base+16..+23}
    A.q[0] = *reinterpret_cast<const uint4*>(ap);
    A.q[1] = *reinterpret_cast<const uint4*>(ap + 16);
    const __bf16* bp = bcol + kb * 32 + sel * 8;     // K {base..+7, base+16..+23} of col o
    Bm.q[0] = *reinterpret_cast<const uint4*>(bp);
    Bm.q[1] = *reinterpret_cast<const uint4*>(bp + 16);
    acc = __builtin_amdgcn_wmma_f32_16x16x32_bf16(false, A.v, false, Bm.v,
                                                  (short)0, acc, false, false);
  }
  float bias = b_lin[o];
#pragma unroll
  for (int r = 0; r < 8; ++r) {  // D: VGPR r -> point row r + 8*sel
    float v = leakyf(acc[r] + bias);
    outF[((size_t)b * Cout + o) * NN + (tileN + sel * 8 + r)] = v;
  }
}

// ---------------- fused mlp1 + mlp2 + last ---------------------------------
__global__ __launch_bounds__(256) void mlp_kernel(
    const float* __restrict__ x, const __bf16* __restrict__ w1,
    const float* __restrict__ b1, const __bf16* __restrict__ w2,
    const float* __restrict__ b2, const float* __restrict__ wlast,
    const float* __restrict__ blast, float* __restrict__ fOut,
    float* __restrict__ flowOut) {
  __shared__ __bf16 XL[16 * 128];
  __shared__ __bf16 H1[16 * 128];
  __shared__ float  H2[16 * 64];

  const int t     = threadIdx.x;
  const int tile  = blockIdx.x;
  const int b     = tile / (NN / 16);
  const int tileN = (tile % (NN / 16)) * 16;

  for (int i = t; i < 16 * 128; i += 256) {   // coalesced over point index
    int p = i & 15;
    int c = i >> 4;
    XL[p * 128 + c] = tobf(x[((size_t)b * 128 + c) * NN + tileN + p]);
  }
  __syncthreads();

  const int lane = t & 31;
  const int wid  = t >> 5;
  const int rc   = lane & 15;
  const int sel  = lane >> 4;

  // mlp1: H1[16 x 128] = leaky(XL * W1^T + b1)
  {
    const int o = wid * 16 + rc;
    v8f acc = {};
    const __bf16* arow = XL + rc * 128;
    const __bf16* bcol = w1 + (size_t)o * 128;
#pragma unroll
    for (int kb = 0; kb < 4; ++kb) {
      FragBF A, Bm;
      const __bf16* ap = arow + kb * 32 + sel * 8;
      A.q[0] = *reinterpret_cast<const uint4*>(ap);
      A.q[1] = *reinterpret_cast<const uint4*>(ap + 16);
      const __bf16* bp = bcol + kb * 32 + sel * 8;
      Bm.q[0] = *reinterpret_cast<const uint4*>(bp);
      Bm.q[1] = *reinterpret_cast<const uint4*>(bp + 16);
      acc = __builtin_amdgcn_wmma_f32_16x16x32_bf16(false, A.v, false, Bm.v,
                                                    (short)0, acc, false, false);
    }
    float bias = b1[o];
#pragma unroll
    for (int r = 0; r < 8; ++r) {
      float v = leakyf(acc[r] + bias);
      H1[(sel * 8 + r) * 128 + o] = tobf(v);
    }
  }
  __syncthreads();

  // mlp2: waves 0..3 produce the 64 flow-feature channels
  if (wid < 4) {
    const int o = wid * 16 + rc;
    v8f acc = {};
    const __bf16* arow = H1 + rc * 128;
    const __bf16* bcol = w2 + (size_t)o * 128;
#pragma unroll
    for (int kb = 0; kb < 4; ++kb) {
      FragBF A, Bm;
      const __bf16* ap = arow + kb * 32 + sel * 8;
      A.q[0] = *reinterpret_cast<const uint4*>(ap);
      A.q[1] = *reinterpret_cast<const uint4*>(ap + 16);
      const __bf16* bp = bcol + kb * 32 + sel * 8;
      Bm.q[0] = *reinterpret_cast<const uint4*>(bp);
      Bm.q[1] = *reinterpret_cast<const uint4*>(bp + 16);
      acc = __builtin_amdgcn_wmma_f32_16x16x32_bf16(false, A.v, false, Bm.v,
                                                    (short)0, acc, false, false);
    }
    float bias = b2[o];
#pragma unroll
    for (int r = 0; r < 8; ++r) {
      float v = leakyf(acc[r] + bias);
      int   p = sel * 8 + r;
      H2[p * 64 + o] = v;
      fOut[((size_t)b * 64 + o) * NN + tileN + p] = v;  // output 1: flow_feat
    }
  }
  __syncthreads();

  // last: flow[3] = W_last(3x64) * h2 + b_last  (tiny, VALU)
  if (t < 48) {
    int   p = t / 3, o = t - p * 3;
    float acc = blast[o];
    for (int c = 0; c < 64; ++c)
      acc = fmaf(wlast[o * 64 + c], H2[p * 64 + c], acc);
    flowOut[((size_t)b * 3 + o) * NN + tileN + p] = acc;  // output 2: flow
  }
}

// ---------------- host launcher --------------------------------------------
extern "C" void kernel_launch(void* const* d_in, const int* in_sizes, int n_in,
                              void* d_out, int out_size, void* d_ws, size_t ws_size,
                              hipStream_t stream) {
  const float* xyz    = (const float*)d_in[0];
  const float* feat   = (const float*)d_in[1];
  const int*   knn    = (const int*)d_in[2];
  const float* w_wn1  = (const float*)d_in[3];
  const float* b_wn1  = (const float*)d_in[4];
  const float* w_lin1 = (const float*)d_in[5];
  const float* b_lin1 = (const float*)d_in[6];
  const float* w_wn2  = (const float*)d_in[7];
  const float* b_wn2  = (const float*)d_in[8];
  const float* w_lin2 = (const float*)d_in[9];
  const float* b_lin2 = (const float*)d_in[10];
  const float* w_mlp1 = (const float*)d_in[11];
  const float* b_mlp1 = (const float*)d_in[12];
  const float* w_mlp2 = (const float*)d_in[13];
  const float* b_mlp2 = (const float*)d_in[14];
  const float* w_last = (const float*)d_in[15];
  const float* b_last = (const float*)d_in[16];

  // workspace layout (bytes)
  char*   ws   = (char*)d_ws;
  __bf16* wbf1 = (__bf16*)(ws + 0);                  // 128*2080*2 = 532480
  __bf16* wbf2 = (__bf16*)(ws + 532480);             // 128*1056*2 = 270336
  __bf16* wm1  = (__bf16*)(ws + 802816);             // 128*128*2  =  32768
  __bf16* wm2  = (__bf16*)(ws + 835584);             // 64*128*2   =  16384
  float*  f1   = (float*)(ws + 851968);              // B*128*N*4  = 16 MB
  float*  f2   = (float*)(ws + 851968 + 16777216);   // B*128*N*4  = 16 MB

  cvt_pad_bf16<<<256, 256, 0, stream>>>(w_lin1, wbf1, 128, 2072, 2080);
  cvt_pad_bf16<<<128, 256, 0, stream>>>(w_lin2, wbf2, 128, 1048, 1056);
  cvt_pad_bf16<<<64, 256, 0, stream>>>(w_mlp1, wm1, 128, 128, 128);
  cvt_pad_bf16<<<32, 256, 0, stream>>>(w_mlp2, wm2, 64, 128, 128);

  // layer-1 tile needs 75,776 B of dynamic LDS (> 64 KB default cap)
  (void)hipFuncSetAttribute(reinterpret_cast<const void*>(point_conv_kernel),
                            hipFuncAttributeMaxDynamicSharedMemorySize, 76800);

  dim3 grid(BB * (NN / 16));
  point_conv_kernel<<<grid, 256, 1024 + 8192 + 16 * 2080 * 2, stream>>>(
      xyz, feat, knn, w_wn1, b_wn1, wbf1, b_lin1, f1, 256, 2080);
  point_conv_kernel<<<grid, 256, 1024 + 8192 + 16 * 1056 * 2, stream>>>(
      xyz, f1, knn, w_wn2, b_wn2, wbf2, b_lin2, f2, 128, 1056);

  float* fOut    = (float*)d_out;
  float* flowOut = fOut + (size_t)BB * 64 * NN;
  mlp_kernel<<<grid, 256, 0, stream>>>(f2, wm1, b_mlp1, wm2, b_mlp2,
                                       w_last, b_last, fOut, flowOut);
}
